// Model_39676907884250
// MI455X (gfx1250) — matrix operand order repaired and checked by reading, not compiled, not run on hardware
//
#include <hip/hip_runtime.h>

#define B_   2
#define H_   16
#define SQ_  2048
#define SK_  2048
#define D_   64
#define DV_  64
#define KT_  32          // kv tile size
#define QB_  128         // q rows per workgroup (8 waves * 16)
#define NWAVE 8
#define THREADS 256
#define NIT_  (SK_ / KT_)

#define KSTRIDE 72       // halfs per Kt row (64 + 8 pad) -> 144B, 16B-aligned rows
#define VSTRIDE 40       // halfs per Vt row (32 + 8 pad) -> 80B
#define PSTRIDE 40       // halfs per P  row (32 + 8 pad) -> 80B

typedef __attribute__((ext_vector_type(16))) __bf16        v16bf;
typedef __attribute__((ext_vector_type(2)))  __bf16        v2bf;
typedef __attribute__((ext_vector_type(8)))  float         v8f;
typedef __attribute__((ext_vector_type(4)))  float         v4f;
typedef __attribute__((ext_vector_type(16))) unsigned short v16u;
typedef __attribute__((ext_vector_type(8)))  unsigned short v8u;
typedef __attribute__((ext_vector_type(2)))  unsigned int  v2ui;
typedef __attribute__((ext_vector_type(4)))  int           v4i;

// ---- gfx1250 async global->LDS path (guarded; falls back cleanly) ----
#if defined(__has_builtin)
#  if __has_builtin(__builtin_amdgcn_global_load_async_to_lds_b128) && \
      __has_builtin(__builtin_amdgcn_s_wait_asynccnt)
#    define USE_ASYNC 1
#  endif
#  if __has_builtin(__builtin_amdgcn_cvt_pk_bf16_f32)
#    define HAVE_PKBF 1
#  endif
#endif
#ifndef USE_ASYNC
#  define USE_ASYNC 0
#endif
#ifndef HAVE_PKBF
#  define HAVE_PKBF 0
#endif
#define AS1 __attribute__((address_space(1)))
#define AS3 __attribute__((address_space(3)))
typedef AS1 v4i* gvec_t;   // global 16B vector pointer (param 1 of async builtin)
typedef AS3 v4i* lvec_t;   // LDS    16B vector pointer (param 2 of async builtin)

__device__ __forceinline__ unsigned short f2bf(float f) {
    union { float f; unsigned int u; } x; x.f = f;
    unsigned int r = x.u + 0x7FFFu + ((x.u >> 16) & 1u);   // RNE
    return (unsigned short)(r >> 16);
}
__device__ __forceinline__ unsigned int pk2bf(float a, float b) {   // {lo=a, hi=b}
#if HAVE_PKBF
    return __builtin_bit_cast(unsigned int, __builtin_amdgcn_cvt_pk_bf16_f32(a, b));
#else
    return (unsigned int)f2bf(a) | ((unsigned int)f2bf(b) << 16);
#endif
}
__device__ __forceinline__ unsigned int hash32(unsigned int x) {
    x ^= x >> 16; x *= 0x7feb352dU; x ^= x >> 15; x *= 0x846ca68bU; x ^= x >> 16;
    return x;
}
__device__ __forceinline__ v16u cat8(v8u a, v8u b) {
    v16u r;
#pragma unroll
    for (int i = 0; i < 8; ++i) { r[i] = a[i]; r[8 + i] = b[i]; }
    return r;
}
__device__ __forceinline__ v8f wmma_bf16(v16u a, v16u b, v8f c) {
    return __builtin_amdgcn_wmma_f32_16x16x32_bf16(
        false, __builtin_bit_cast(v16bf, a),
        false, __builtin_bit_cast(v16bf, b),
        (short)0, c, false, false);
}

__global__ __launch_bounds__(THREADS)
void fa_wmma_kernel(const float* __restrict__ q, const float* __restrict__ k,
                    const float* __restrict__ v, float* __restrict__ out)
{
    // double-buffered staging + compute tiles (~62 KB of the 320 KB WGP LDS)
    __shared__ float          ldsKf[2][KT_ * D_];          // fp32 async-DMA dest
    __shared__ float          ldsVf[2][KT_ * DV_];
    __shared__ unsigned short Kt[2][KT_ * KSTRIDE];        // bf16 K tile [kv][d]
    __shared__ unsigned short Vt[2][DV_ * VSTRIDE];        // bf16 V tile transposed [dv][kv]
    __shared__ unsigned short Pt[NWAVE * 16 * PSTRIDE];    // per-wave P transpose scratch

    const int tid  = threadIdx.x;
    const int lane = tid & 31;
    const int wv   = tid >> 5;
    const int hs   = lane >> 4;   // 0: lanes 0-15, 1: lanes 16-31
    const int ln   = lane & 15;

    const int    hh   = blockIdx.y;
    const int    bb   = blockIdx.z;
    const int    q0   = blockIdx.x * QB_ + wv * 16;
    const size_t qoff = (size_t)(bb * H_ + hh) * SQ_;

    const float* kgbase = k + (size_t)hh * SK_ * D_;
    const float* vgbase = v + (size_t)hh * SK_ * DV_;
    unsigned short* Pw = &Pt[wv * 16 * PSTRIDE];

#if USE_ASYNC
    auto stage_load = [&](int buf, int kv0) {               // issue async DMA, no wait
        const float* kg = kgbase + (size_t)kv0 * D_;
        const float* vg = vgbase + (size_t)kv0 * DV_;
#pragma unroll
        for (int i = 0; i < 2; ++i) {
            const int f4 = tid + i * THREADS;               // 512 x b128 = 32x64 fp32
            __builtin_amdgcn_global_load_async_to_lds_b128(
                (gvec_t)(kg + f4 * 4), (lvec_t)&ldsKf[buf][f4 * 4], 0, 0);
            __builtin_amdgcn_global_load_async_to_lds_b128(
                (gvec_t)(vg + f4 * 4), (lvec_t)&ldsVf[buf][f4 * 4], 0, 0);
        }
    };
#endif
    auto tile_convert = [&](int buf, int kv0) {             // fp32 -> bf16 compute tiles
        const float* kg = kgbase + (size_t)kv0 * D_;
        const float* vg = vgbase + (size_t)kv0 * DV_;
        (void)kg; (void)vg;
#pragma unroll
        for (int i = 0; i < 2; ++i) {
            const int f4  = tid + i * THREADS;
            const int e0  = f4 * 4;
            const int row = e0 >> 6;                        // kv within tile
            const int col = e0 & 63;                        // d / dv
#if USE_ASYNC
            v4f kk4 = *(const v4f*)&ldsKf[buf][e0];
            v4f vv4 = *(const v4f*)&ldsVf[buf][e0];
#else
            v4f kk4 = *(const v4f*)(kg + e0);
            v4f vv4 = *(const v4f*)(vg + e0);
#endif
            v2ui kh;
            kh[0] = pk2bf(kk4[0], kk4[1]);
            kh[1] = pk2bf(kk4[2], kk4[3]);
            *(v2ui*)&Kt[buf][row * KSTRIDE + col] = kh;
#pragma unroll
            for (int c4 = 0; c4 < 4; ++c4)
                Vt[buf][(col + c4) * VSTRIDE + row] = f2bf(vv4[c4]);
        }
    };

    // ---- Q strip: fp32 -> bf16, directly in WMMA A-layout (2 chunks of K=32) ----
    v16u qa[2];
    {
        const float* qrow = q + (qoff + (size_t)(q0 + ln)) * D_;
#pragma unroll
        for (int c = 0; c < 2; ++c) {
            const int d0 = c * 32 + hs * 8;
            v4f f0 = *(const v4f*)(qrow + d0);
            v4f f1 = *(const v4f*)(qrow + d0 + 4);
            v4f f2 = *(const v4f*)(qrow + d0 + 16);
            v4f f3 = *(const v4f*)(qrow + d0 + 20);
            v16u t;
#pragma unroll
            for (int i = 0; i < 4; ++i) {
                t[i]      = f2bf(f0[i]);
                t[4 + i]  = f2bf(f1[i]);
                t[8 + i]  = f2bf(f2[i]);
                t[12 + i] = f2bf(f3[i]);
            }
            qa[c] = t;
        }
    }

    const v8f zero8 = {0.f, 0.f, 0.f, 0.f, 0.f, 0.f, 0.f, 0.f};
    v8f oacc[4];
#pragma unroll
    for (int n = 0; n < 4; ++n) oacc[n] = zero8;
    float m[8], l[8];
#pragma unroll
    for (int j = 0; j < 8; ++j) { m[j] = -3.0e38f; l[j] = 0.f; }

    // ---- pipeline prologue: tile 0 staged+converted, tile 1 in flight ----
#if USE_ASYNC
    stage_load(0, 0);
    __builtin_amdgcn_s_wait_asynccnt(0);
    __syncthreads();
#endif
    tile_convert(0, 0);
#if USE_ASYNC
    if (NIT_ > 1) stage_load(1, KT_);
#endif
    __syncthreads();                                        // tiles[0] ready

    const unsigned bh = (unsigned)(bb * H_ + hh);

    for (int it = 0; it < NIT_; ++it) {
        const int cur = it & 1;
        const int kv0 = it * KT_;

        // ---- S = Q * K^T (16x32 tile -> two 16x16 WMMA accumulators) ----
        v8f s[2];
#pragma unroll
        for (int t = 0; t < 2; ++t) {
            v8f acc = zero8;
#pragma unroll
            for (int c = 0; c < 2; ++c) {
                const int kvl  = t * 16 + ln;
                const int dcol = c * 32 + hs * 8;
                const v8u* kp = (const v8u*)&Kt[cur][kvl * KSTRIDE + dcol];
                acc = wmma_bf16(qa[c], cat8(kp[0], kp[2]), acc);
            }
            s[t] = acc;
        }

        // ---- online softmax + dropout + write P (bf16) to LDS ----
#pragma unroll
        for (int j = 0; j < 8; ++j) {
            float tm = fmaxf(s[0][j], s[1][j]);
#pragma unroll
            for (int off = 8; off >= 1; off >>= 1)
                tm = fmaxf(tm, __shfl_xor(tm, off, 32));
            const float mn = fmaxf(m[j], tm);
            const float sc = exp2f((m[j] - mn) * 1.44269504f);
            m[j] = mn;
            float p0 = exp2f((s[0][j] - mn) * 1.44269504f);
            float p1 = exp2f((s[1][j] - mn) * 1.44269504f);
            float rs = p0 + p1;
#pragma unroll
            for (int off = 8; off >= 1; off >>= 1)
                rs += __shfl_xor(rs, off, 32);
            l[j] = l[j] * sc + rs;                          // denom BEFORE dropout
            oacc[0][j] *= sc; oacc[1][j] *= sc;
            oacc[2][j] *= sc; oacc[3][j] *= sc;

            // deterministic dropout (seed 42): one hash -> two 16-bit draws
            const unsigned qrow = (unsigned)(q0 + j + 8 * hs);
            const unsigned hsh  =
                hash32(((bh << 22) | (qrow << 11) | (unsigned)(kv0 + ln)) ^ 0x2Au);
            if ((hsh & 0xffffu) >= 58982u) p0 = 0.f;        // keep-prob 0.9
            if ((hsh >> 16)     >= 58982u) p1 = 0.f;

            Pw[(j + 8 * hs) * PSTRIDE + ln]      = f2bf(p0);
            Pw[(j + 8 * hs) * PSTRIDE + 16 + ln] = f2bf(p1);
        }

        // ---- O += P * V ----
        v8u a0 = *(const v8u*)&Pw[ln * PSTRIDE + hs * 8];
        v8u a1 = *(const v8u*)&Pw[ln * PSTRIDE + 16 + hs * 8];
        v16u pa = cat8(a0, a1);
#pragma unroll
        for (int n = 0; n < 4; ++n) {
            const int dv = n * 16 + ln;
            const v8u* vp = (const v8u*)&Vt[cur][dv * VSTRIDE + hs * 8];
            oacc[n] = wmma_bf16(pa, cat8(vp[0], vp[2]), oacc[n]);
        }

        // ---- pipeline: convert next tile, prefetch tile after next ----
        if (it + 1 < NIT_) {
#if USE_ASYNC
            __builtin_amdgcn_s_wait_asynccnt(0);            // stage[nxt] landed
#endif
            __syncthreads();                                // all waves past compute
            tile_convert(cur ^ 1, kv0 + KT_);
#if USE_ASYNC
            if (it + 2 < NIT_) stage_load(cur, kv0 + 2 * KT_);
#endif
            __syncthreads();                                // tiles[nxt] ready
        }
    }

    // ---- epilogue: fold post-softmax 0.5 and dropout 1/0.9 into 1/l ----
#pragma unroll
    for (int j = 0; j < 8; ++j) {
        const float norm = (0.5f / 0.9f) / l[j];
        const int row = q0 + j + 8 * hs;
        float* op = out + (qoff + (size_t)row) * DV_;
#pragma unroll
        for (int n = 0; n < 4; ++n)
            op[n * 16 + ln] = oacc[n][j] * norm;
    }
}

extern "C" void kernel_launch(void* const* d_in, const int* in_sizes, int n_in,
                              void* d_out, int out_size, void* d_ws, size_t ws_size,
                              hipStream_t stream) {
    (void)in_sizes; (void)n_in; (void)out_size; (void)d_ws; (void)ws_size;
    const float* q = (const float*)d_in[0];
    const float* k = (const float*)d_in[1];
    const float* v = (const float*)d_in[2];
    float*       o = (float*)d_out;
    dim3 grid(SQ_ / QB_, H_, B_);   // (16, 16, 2)
    fa_wmma_kernel<<<grid, THREADS, 0, stream>>>(q, k, v, o);
}
